// CustomBertSelfAttention_6459630814126
// MI455X (gfx1250) — matrix operand order repaired and checked by reading, not compiled, not run on hardware
//
#include <hip/hip_runtime.h>

// ---------------------------------------------------------------------------
// BERT self-attention for gfx1250 (MI455X).  B=16, S=512, H=768, NH=12, HD=64.
// All matmuls via v_wmma_f32_16x16x32_f16.  Memory-bound on the 201MB probs
// output, so everything else stays in f16 workspace / LDS.  V-transpose is
// staged into LDS with the Tensor Data Mover (tensor_load_to_lds).
// ---------------------------------------------------------------------------

typedef _Float16 v16h __attribute__((ext_vector_type(16)));
typedef float    v8f  __attribute__((ext_vector_type(8)));
typedef float    v4f  __attribute__((ext_vector_type(4)));
typedef int      v4i  __attribute__((ext_vector_type(4)));
typedef int      v8i  __attribute__((ext_vector_type(8)));

#define BB   16
#define SS   512
#define HH   768
#define NHD  12
#define HDD  64

union FragU { v16h h; v4f f[2]; };

// A fragment, 16x32 f16 (MxK).  rowbase = &mat[row][kbase], row k-contiguous.
// ISA layout: lanes 0-15 hold K 0..7 (v0-3) and 16..23 (v4-7); lanes 16-31
// hold K 8..15 and 24..31.
__device__ __forceinline__ v16h load_frag_a(const _Float16* rowbase, int lane) {
  const int klo = (lane < 16) ? 0 : 8;
  FragU u;
  u.f[0] = *reinterpret_cast<const v4f*>(rowbase + klo);
  u.f[1] = *reinterpret_cast<const v4f*>(rowbase + klo + 16);
  return u.h;
}

// B fragment, 32x16 f16 (KxN), column n = lane%16 stored as a k-contiguous
// row in memory (i.e. memory holds B^T rows).  Lanes 0-15: K 0..7 and 16..23,
// lanes 16-31: K 8..15 and 24..31, 2 halves per VGPR.
__device__ __forceinline__ v16h load_frag_b(const _Float16* rowbase, int lane) {
  const int klo = (lane < 16) ? 0 : 8;
  FragU u;
  u.f[0] = *reinterpret_cast<const v4f*>(rowbase + klo);
  u.f[1] = *reinterpret_cast<const v4f*>(rowbase + klo + 16);
  return u.h;
}

__device__ __forceinline__ v8f wmma16(v16h a, v16h b, v8f c) {
  return __builtin_amdgcn_wmma_f32_16x16x32_f16(false, a, false, b,
                                                (short)0, c, false, false);
}

#define ZERO8 {0.f,0.f,0.f,0.f,0.f,0.f,0.f,0.f}

// ---------------------------------------------------------------------------
__global__ void convert_f16(const float* __restrict__ src,
                            _Float16* __restrict__ dst, int n) {
  int i = blockIdx.x * blockDim.x + threadIdx.x;
  if (i < n) dst[i] = (_Float16)src[i];
}

// ---------------------------------------------------------------------------
// y = x @ W^T + bias, written as f16 in head layout [B,NH,S,HD].
// One 16x16 output tile per wave; K=768 -> 24 WMMA steps.
__global__ __launch_bounds__(256) void qkv_gemm(
    const _Float16* __restrict__ X,   // [8192, 768]
    const _Float16* __restrict__ W,   // [768, 768] rows k-contiguous
    const float*    __restrict__ bias,
    _Float16*       __restrict__ out) // [B*NH, S, HD]
{
  const int lane = threadIdx.x & 31;
  const int wid  = blockIdx.x * 8 + (threadIdx.x >> 5);
  const int NT   = HH / 16;             // 48 n-tiles
  const int mt   = wid / NT;
  const int nt   = wid % NT;
  const int m0   = mt * 16, n0 = nt * 16;

  const _Float16* xrow = X + (long)(m0 + (lane & 15)) * HH;
  const _Float16* wrow = W + (long)(n0 + (lane & 15)) * HH;

  v8f c = ZERO8;
  for (int k = 0; k < HH; k += 32) {
    __builtin_prefetch(xrow + k + 64, 0, 3);
    __builtin_prefetch(wrow + k + 64, 0, 3);
    v16h a = load_frag_a(xrow + k, lane);
    v16h b = load_frag_b(wrow + k, lane);
    c = wmma16(a, b, c);
  }

  const float bv = bias[n0 + (lane & 15)];
  const int h = n0 / HDD;
  const int d = (n0 % HDD) + (lane & 15);
#pragma unroll
  for (int r = 0; r < 8; ++r) {
    const int m  = m0 + r + ((lane >> 4) << 3);  // C layout: M = r + 8*(lane/16)
    const int b_ = m >> 9;                       // batch
    const int s  = m & 511;                      // seq
    const long o = (((long)(b_ * NHD + h) * SS + s) * HDD) + d;
    out[o] = (_Float16)(c[r] + bv);
  }
}

// ---------------------------------------------------------------------------
// V [BH, 512, 64] -> Vt [BH, 64, 512] so probs@V B-fragments are row reads.
// The 64x64 f16 input tile is DMA'd into LDS by the Tensor Data Mover with
// hardware row padding (4 DWORDs pad per 32 DWORDs -> LDS pitch 72 halves),
// replacing 4096 scalar global loads per block with one tensor_load_to_lds.
__global__ __launch_bounds__(256) void transpose_v(
    const _Float16* __restrict__ V, _Float16* __restrict__ Vt)
{
  __shared__ _Float16 t[64][72];
  const int bh = blockIdx.x >> 3;
  const int s0 = (blockIdx.x & 7) * 64;
  const _Float16* vin  = V  + ((long)bh * SS + s0) * HDD;
  _Float16*       vout = Vt + (long)bh * HDD * SS + s0;

  if (threadIdx.x < 32) {             // one wave owns the TDM op (EXEC ignored)
    const unsigned long long ga = (unsigned long long)vin;
    const unsigned lds = (unsigned)(unsigned long long)&t[0][0]; // low 32 = LDS offset

    // D# group 0: count=1 | lds_addr | global_addr[56:0] | type=2
    v4i g0;
    g0.x = 1;
    g0.y = (int)lds;
    g0.z = (int)(ga & 0xffffffffu);
    g0.w = (int)(((ga >> 32) & 0x01ffffffu) | 0x80000000u);

    // D# group 1: data_size=2B | pad_enable | pad_interval=32dw | pad_amount=4dw
    //             tensor 64 x 512 (stride0=64, stride1=32768), tile 64 x 64
    v8i g1;
    g1.s0 = (1 << 16) | (1 << 20) | (4 << 22) | (3 << 25);
    g1.s1 = (64  << 16);                 // tensor_dim0[15:0] = 64
    g1.s2 = (512 << 16);                 // tensor_dim1[15:0] = 512
    g1.s3 = (64  << 16);                 // tile_dim0 = 64
    g1.s4 = 64;                          // tile_dim1 = 64, tile_dim2 = 0
    g1.s5 = 64;                          // tensor_dim0_stride[31:0] = 64
    g1.s6 = (int)0x80000000u;            // tensor_dim1_stride[15:0] = 32768
    g1.s7 = 0;                           // tensor_dim1_stride[47:16] = 0

    asm volatile("tensor_load_to_lds %0, %1" :: "s"(g0), "s"(g1) : "memory");
    __builtin_amdgcn_s_wait_tensorcnt(0);
  }
  __syncthreads();

  for (int i = threadIdx.x; i < 64 * 64; i += 256) {
    const int dd = i >> 6, sc = i & 63;
    vout[dd * SS + sc] = t[sc][dd];
  }
}

// ---------------------------------------------------------------------------
// Fused attention: one 16-row query tile per wave, 2 waves per block.
// Pass A: scores + row max.  Pass B: recompute scores, exp, row sum,
// stash e (f16) in LDS.  Then: coalesced probs store, and ctx = e @ Vt
// via WMMA with A-fragments read back from LDS.
__global__ __launch_bounds__(64) void attention(
    const _Float16* __restrict__ Q,    // [BH, S, HD]
    const _Float16* __restrict__ K,    // [BH, S, HD]
    const _Float16* __restrict__ Vt,   // [BH, HD, S]
    const float*    __restrict__ mask, // [B, S]
    float*          __restrict__ ctx,  // [B, S, H]
    float*          __restrict__ probs)// [BH, S, S]
{
  __shared__ _Float16 es[2][16][SS];     // 32 KB: exp(scores - max), f16
  __shared__ float    rinv_s[2][16];

  const int lane = threadIdx.x & 31;
  const int wave = threadIdx.x >> 5;
  const int bh   = blockIdx.x >> 4;
  const int qblk = blockIdx.x & 15;
  const int b    = bh / NHD, h = bh % NHD;
  const int m0   = (qblk * 2 + wave) * 16;
  const float scale = 0.125f;            // 1/sqrt(64)

  const _Float16* Qb = Q  + (long)bh * SS * HDD;
  const _Float16* Kb = K  + (long)bh * SS * HDD;
  const _Float16* Vb = Vt + (long)bh * HDD * SS;
  const float*  mrow = mask + b * SS;

  const _Float16* qrow = Qb + (long)(m0 + (lane & 15)) * HDD;
  const v16h a0 = load_frag_a(qrow, lane);
  const v16h a1 = load_frag_a(qrow + 32, lane);

  // ---- pass A: row max over 512 keys -------------------------------------
  float rmax[8];
#pragma unroll
  for (int r = 0; r < 8; ++r) rmax[r] = -3.0e38f;

  for (int j = 0; j < 32; ++j) {
    const _Float16* krow = Kb + (long)(j * 16 + (lane & 15)) * HDD;
    __builtin_prefetch(krow + 16 * HDD, 0, 3);
    v16h b0 = load_frag_b(krow, lane);
    v16h b1 = load_frag_b(krow + 32, lane);
    v8f c = ZERO8;
    c = wmma16(a0, b0, c);
    c = wmma16(a1, b1, c);
    const float madd = (1.0f - mrow[j * 16 + (lane & 15)]) * -10000.0f;
#pragma unroll
    for (int r = 0; r < 8; ++r)
      rmax[r] = fmaxf(rmax[r], c[r] * scale + madd);
  }
  // reduce across the 16 lanes holding the same C-row (xor 1,2,4,8 stays in group)
#pragma unroll
  for (int off = 8; off >= 1; off >>= 1)
#pragma unroll
    for (int r = 0; r < 8; ++r)
      rmax[r] = fmaxf(rmax[r], __shfl_xor(rmax[r], off, 32));

  // ---- pass B: exp, row sum, stash e in LDS ------------------------------
  float rsum[8];
#pragma unroll
  for (int r = 0; r < 8; ++r) rsum[r] = 0.0f;

  for (int j = 0; j < 32; ++j) {
    const _Float16* krow = Kb + (long)(j * 16 + (lane & 15)) * HDD;
    v16h b0 = load_frag_b(krow, lane);
    v16h b1 = load_frag_b(krow + 32, lane);
    v8f c = ZERO8;
    c = wmma16(a0, b0, c);
    c = wmma16(a1, b1, c);
    const float madd = (1.0f - mrow[j * 16 + (lane & 15)]) * -10000.0f;
#pragma unroll
    for (int r = 0; r < 8; ++r) {
      const float e = __expf(c[r] * scale + madd - rmax[r]);
      rsum[r] += e;
      const int mm = r + ((lane >> 4) << 3);
      es[wave][mm][j * 16 + (lane & 15)] = (_Float16)e;
    }
  }
#pragma unroll
  for (int off = 8; off >= 1; off >>= 1)
#pragma unroll
    for (int r = 0; r < 8; ++r)
      rsum[r] += __shfl_xor(rsum[r], off, 32);

  float rinv[8];                          // matches C layout: row r + 8*(lane/16)
#pragma unroll
  for (int r = 0; r < 8; ++r) rinv[r] = 1.0f / rsum[r];

  if ((lane & 15) == 0) {
    const int base = (lane >> 4) << 3;
#pragma unroll
    for (int r = 0; r < 8; ++r) rinv_s[wave][base + r] = rinv[r];
  }

  // ---- probs store: coalesced 128B f32 stores (dominant HBM traffic) -----
  float* pout = probs + ((long)bh * SS + m0) * SS;
  for (int i = lane; i < 16 * SS; i += 32) {
    const int mm = i >> 9, col = i & 511;
    pout[i] = (float)es[wave][mm][col] * rinv_s[wave][mm];
  }

  // ---- ctx = (e @ Vt^T_rows) * rinv --------------------------------------
  v8f acc[4] = {ZERO8, ZERO8, ZERO8, ZERO8};
  for (int ks = 0; ks < 16; ++ks) {
    const _Float16* erow = &es[wave][lane & 15][ks * 32];
    v16h a = load_frag_a(erow, lane);        // ds_load_b128 from LDS
#pragma unroll
    for (int nt = 0; nt < 4; ++nt) {
      const _Float16* vrow = Vb + (long)(nt * 16 + (lane & 15)) * SS + ks * 32;
      v16h bb = load_frag_b(vrow, lane);
      acc[nt] = wmma16(a, bb, acc[nt]);
    }
  }

#pragma unroll
  for (int nt = 0; nt < 4; ++nt) {
#pragma unroll
    for (int r = 0; r < 8; ++r) {
      const int mm = m0 + r + ((lane >> 4) << 3);
      const long o = ((long)(b * SS + mm) * HH) + h * HDD + nt * 16 + (lane & 15);
      ctx[o] = acc[nt][r] * rinv[r];
    }
  }
}

// ---------------------------------------------------------------------------
extern "C" void kernel_launch(void* const* d_in, const int* in_sizes, int n_in,
                              void* d_out, int out_size, void* d_ws, size_t ws_size,
                              hipStream_t stream) {
  const float* hidden = (const float*)d_in[0];
  const float* amask  = (const float*)d_in[1];
  const float* Wq     = (const float*)d_in[2];
  const float* bq     = (const float*)d_in[3];
  const float* Wk     = (const float*)d_in[4];
  const float* bk     = (const float*)d_in[5];
  const float* Wv     = (const float*)d_in[6];
  const float* bv     = (const float*)d_in[7];

  float* ctx   = (float*)d_out;
  float* probs = ctx + (long)BB * SS * HH;   // tuple: (ctx, probs) flat

  const long nH = (long)BB * SS * HH;        // 6,291,456
  const long nW = (long)HH * HH;             //   589,824
  _Float16* ws   = (_Float16*)d_ws;
  _Float16* h16  = ws;
  _Float16* wq16 = h16  + nH;
  _Float16* wk16 = wq16 + nW;
  _Float16* wv16 = wk16 + nW;
  _Float16* q16  = wv16 + nW;
  _Float16* k16  = q16  + nH;
  _Float16* v16  = k16  + nH;
  _Float16* vt16 = v16  + nH;                // ~66.5 MB total workspace

  convert_f16<<<(int)((nH + 255) / 256), 256, 0, stream>>>(hidden, h16, (int)nH);
  convert_f16<<<(int)((nW + 255) / 256), 256, 0, stream>>>(Wq, wq16, (int)nW);
  convert_f16<<<(int)((nW + 255) / 256), 256, 0, stream>>>(Wk, wk16, (int)nW);
  convert_f16<<<(int)((nW + 255) / 256), 256, 0, stream>>>(Wv, wv16, (int)nW);

  // 512 m-tiles * 48 n-tiles = 24576 waves, 8 waves per 256-thread block
  qkv_gemm<<<3072, 256, 0, stream>>>(h16, wq16, bq, q16);
  qkv_gemm<<<3072, 256, 0, stream>>>(h16, wk16, bk, k16);
  qkv_gemm<<<3072, 256, 0, stream>>>(h16, wv16, bv, v16);

  transpose_v<<<192 * 8, 256, 0, stream>>>(v16, vt16);

  // 192 (b,h) * 16 q-blocks; 2 waves (two 16-row q tiles) per block
  attention<<<192 * 16, 64, 0, stream>>>(q16, k16, vt16, amask, ctx, probs);
}
